// RNN_21801253995070
// MI455X (gfx1250) — matrix operand order repaired and checked by reading, not compiled, not run on hardware
//
#include <hip/hip_runtime.h>

namespace {

constexpr int B = 64, S = 512, E = 512, H = 1024, O = 1024;
constexpr int NWG_SCAN = H / 64;  // 16 persistent workgroups for the scan

typedef __attribute__((ext_vector_type(16))) __bf16 v16bf;
typedef __attribute__((ext_vector_type(8)))  __bf16 v8bf;
typedef __attribute__((ext_vector_type(4)))  __bf16 v4bf;
typedef __attribute__((ext_vector_type(8)))  float  v8f;

// ---------------- f32 -> bf16 pre-conversion (one pass per operand) ---------
__global__ __launch_bounds__(256) void cvt_f32_bf16_kernel(
    const float* __restrict__ in, __bf16* __restrict__ out, int n4) {
  int i = blockIdx.x * blockDim.x + threadIdx.x;
  if (i < n4) {
    float4 f = *(const float4*)(in + (size_t)i * 4);
    v4bf o;
    o[0] = (__bf16)f.x; o[1] = (__bf16)f.y; o[2] = (__bf16)f.z; o[3] = (__bf16)f.w;
    *(v4bf*)(out + (size_t)i * 4) = o;
  }
}

__global__ void zero_bar_kernel(unsigned int* bar) { bar[0] = 0u; }

// A fragment (16x32 bf16, M x K). Lane L holds row M = L%16.
// Low lanes: K = k0 + 0..15; high lanes: K = k0 + 16..31.
__device__ inline v16bf load_a_frag(const __bf16* __restrict__ A, int row, int ldk,
                                    int k0, int lane) {
  const __bf16* p = A + (size_t)row * ldk + k0 + ((lane >= 16) ? 16 : 0);
  v8bf lo = *(const v8bf*)p;
  v8bf hi = *(const v8bf*)(p + 8);
  return __builtin_shufflevector(lo, hi, 0, 1, 2, 3, 4, 5, 6, 7,
                                 8, 9, 10, 11, 12, 13, 14, 15);
}

// B fragment (32x16 bf16, K x N); weight stored [N,K] row-major. Lane L holds
// N = L%16; low lanes K=k0+0..15, high lanes K=k0+16..31. Works for global or LDS.
__device__ inline v16bf load_b_frag(const __bf16* __restrict__ Bm, int col, int ldk,
                                    int k0, int lane) {
  const __bf16* p = Bm + (size_t)col * ldk + k0 + ((lane >= 16) ? 16 : 0);
  return *(const v16bf*)p;
}

__device__ inline v8f wmma_bf16(v16bf a, v16bf b, v8f c) {
  return __builtin_amdgcn_wmma_f32_16x16x32_bf16(false, a, false, b, (short)0, c,
                                                 false, false);
}

// ---------------- Parallel GEMM for phases 1 & 3 ----------------------------
// C = A(MxK bf16 row-major) * W(NxK bf16 row-major)^T + bias
// MODE 0: (float out)  out[m*N+n] = C + bias[n] + bias2[n]        (wx)
// MODE 2: (float out)  out[((m%B)*S + m/B)*N + n] = C + bias[n]   (y)
// 128 threads = 4 waves (2x2); WG tile 64x128; wave tile 32x64 (2x4 WMMA).
template <int MODE>
__global__ __launch_bounds__(128) void gemm_wmma_kernel(
    const __bf16* __restrict__ A, const __bf16* __restrict__ Bm,
    const float* __restrict__ bias, const float* __restrict__ bias2,
    float* __restrict__ out, int M, int N, int K) {
  const int lane   = threadIdx.x & 31;
  const int wave   = threadIdx.x >> 5;
  const int m_wave = blockIdx.x * 64 + (wave & 1) * 32;
  const int n_wave = blockIdx.y * 128 + (wave >> 1) * 64;
  const int lrow   = lane & 15;

  v8f c[2][4] = {};

  for (int k0 = 0; k0 < K; k0 += 32) {
    v16bf a0 = load_a_frag(A, m_wave + lrow,      K, k0, lane);
    v16bf a1 = load_a_frag(A, m_wave + 16 + lrow, K, k0, lane);
#pragma unroll
    for (int j = 0; j < 4; ++j) {
      v16bf b = load_b_frag(Bm, n_wave + j * 16 + lrow, K, k0, lane);
      c[0][j] = wmma_bf16(a0, b, c[0][j]);
      c[1][j] = wmma_bf16(a1, b, c[1][j]);
    }
  }

  // C/D layout: lane L -> N = L%16; VGPR v -> M = v + 8*(L>=16).
#pragma unroll
  for (int i = 0; i < 2; ++i) {
#pragma unroll
    for (int j = 0; j < 4; ++j) {
      const int nn = n_wave + j * 16 + lrow;
      float bs = bias[nn];
      if (MODE == 0) bs += bias2[nn];
#pragma unroll
      for (int v = 0; v < 8; ++v) {
        const int mm = m_wave + i * 16 + v + ((lane >= 16) ? 8 : 0);
        float val = c[i][j][v] + bs;
        size_t oidx;
        if (MODE == 2) {
          oidx = ((size_t)(mm % B) * S + (size_t)(mm / B)) * (size_t)N + nn;
        } else {
          oidx = (size_t)mm * (size_t)N + nn;
        }
        out[oidx] = val;
      }
    }
  }
}

// ---------------- Persistent recurrent scan ---------------------------------
// 16 workgroups, each owns a fixed 64-column slab of h. U slab (64x1024 bf16 =
// 128KB) staged once in LDS; 512 timesteps with a device-scope atomic barrier.
// hs[t] = ReLU(hs[t-1] * U^T + U_b + wx[:,t,:]), stored bf16.
__global__ __launch_bounds__(128) void rnn_scan_kernel(
    const __bf16* __restrict__ h0b,  // [B,H]
    const __bf16* __restrict__ Ub,   // [H,H] bf16
    const float* __restrict__ U_b,   // [H]
    const float* __restrict__ wx,    // [B,S,H] f32
    __bf16* __restrict__ hs,         // [S,B,H] bf16
    unsigned int* __restrict__ bar) {
  extern __shared__ __bf16 Ulds[];  // 64 * H = 128KB
  const int lane   = threadIdx.x & 31;
  const int wave   = threadIdx.x >> 5;
  const int m_wave = (wave & 1) * 32;
  const int n_loc  = (wave >> 1) * 32;
  const int n_wg   = blockIdx.x * 64;
  const int lrow   = lane & 15;

  // Stage this WG's U slab into LDS (16B chunks, cooperative).
  {
    const __bf16* src = Ub + (size_t)n_wg * H;
    const int total8 = (64 * H) / 8;
    for (int idx = threadIdx.x; idx < total8; idx += blockDim.x)
      *(v8bf*)(Ulds + (size_t)idx * 8) = *(const v8bf*)(src + (size_t)idx * 8);
  }
  __syncthreads();

  // Hoist biases for this lane's two N tiles.
  const float bs0 = U_b[n_wg + n_loc + lrow];
  const float bs1 = U_b[n_wg + n_loc + 16 + lrow];

  for (int t = 0; t < S; ++t) {
    const __bf16* hp = (t == 0) ? h0b : hs + (size_t)(t - 1) * B * H;
    v8f c[2][2] = {};
    for (int k0 = 0; k0 < H; k0 += 32) {
      v16bf a0 = load_a_frag(hp, m_wave + lrow,      H, k0, lane);
      v16bf a1 = load_a_frag(hp, m_wave + 16 + lrow, H, k0, lane);
      v16bf b0 = load_b_frag(Ulds, n_loc + lrow,      H, k0, lane);
      v16bf b1 = load_b_frag(Ulds, n_loc + 16 + lrow, H, k0, lane);
      c[0][0] = wmma_bf16(a0, b0, c[0][0]);
      c[0][1] = wmma_bf16(a0, b1, c[0][1]);
      c[1][0] = wmma_bf16(a1, b0, c[1][0]);
      c[1][1] = wmma_bf16(a1, b1, c[1][1]);
    }

    __bf16* ho = hs + (size_t)t * B * H;
    const float* wxt = wx + (size_t)t * H;  // element (b,h): wxt[b*S*H + h]
#pragma unroll
    for (int i = 0; i < 2; ++i) {
#pragma unroll
      for (int j = 0; j < 2; ++j) {
        const int nn = n_wg + n_loc + j * 16 + lrow;
        const float bs = j ? bs1 : bs0;
#pragma unroll
        for (int v = 0; v < 8; ++v) {
          const int mm = m_wave + i * 16 + v + ((lane >= 16) ? 8 : 0);
          float val = c[i][j][v] + bs + wxt[(size_t)mm * S * H + nn];
          val = val > 0.0f ? val : 0.0f;
          ho[(size_t)mm * H + nn] = (__bf16)val;
        }
      }
    }

    // Device-scope step barrier across the 16 workgroups.
    __threadfence();   // release: make hs[t] visible agent-wide
    __syncthreads();
    if (threadIdx.x == 0) {
      __hip_atomic_fetch_add(bar, 1u, __ATOMIC_ACQ_REL, __HIP_MEMORY_SCOPE_AGENT);
      const unsigned int target = (unsigned int)(NWG_SCAN * (t + 1));
      while (__hip_atomic_load(bar, __ATOMIC_ACQUIRE, __HIP_MEMORY_SCOPE_AGENT) <
             target)
        __builtin_amdgcn_s_sleep(1);
    }
    __syncthreads();
    __threadfence();   // acquire: see other slabs of hs[t]
  }
}

inline void cvt(const float* in, __bf16* out, size_t n, hipStream_t s) {
  int n4 = (int)(n / 4);
  cvt_f32_bf16_kernel<<<(n4 + 255) / 256, 256, 0, s>>>(in, out, n4);
}

}  // namespace

extern "C" void kernel_launch(void* const* d_in, const int* in_sizes, int n_in,
                              void* d_out, int out_size, void* d_ws, size_t ws_size,
                              hipStream_t stream) {
  (void)in_sizes; (void)n_in; (void)out_size; (void)ws_size;
  const float* x   = (const float*)d_in[0];  // [B,S,E]
  const float* h0  = (const float*)d_in[1];  // [B,H]
  const float* U_w = (const float*)d_in[2];  // [H,H]
  const float* U_b = (const float*)d_in[3];  // [H]
  const float* W_w = (const float*)d_in[4];  // [H,E]
  const float* W_b = (const float*)d_in[5];  // [H]
  const float* bb  = (const float*)d_in[6];  // [H]
  const float* V_w = (const float*)d_in[7];  // [O,H]
  const float* V_b = (const float*)d_in[8];  // [O]
  float* y = (float*)d_out;                  // [B,S,O] f32

  // Workspace layout (all sub-buffers keep >=32B alignment):
  const size_t nBSH = (size_t)B * S * H;
  float*  wx  = (float*)d_ws;                 // [B,S,H] f32   128 MB
  __bf16* hs  = (__bf16*)(wx + nBSH);         // [S,B,H] bf16   64 MB
  __bf16* xb  = hs + nBSH;                    // [B*S,E] bf16   32 MB
  __bf16* Wb  = xb + (size_t)B * S * E;       // [H,E]  bf16     1 MB
  __bf16* Ub  = Wb + (size_t)H * E;           // [H,H]  bf16     2 MB
  __bf16* Vb  = Ub + (size_t)H * H;           // [O,H]  bf16     2 MB
  __bf16* h0b = Vb + (size_t)O * H;           // [B,H]  bf16   128 KB
  unsigned int* bar = (unsigned int*)(h0b + (size_t)B * H);

  // One-shot operand conversion to bf16 + barrier init.
  cvt(x,   xb,  (size_t)B * S * E, stream);
  cvt(W_w, Wb,  (size_t)H * E,     stream);
  cvt(U_w, Ub,  (size_t)H * H,     stream);
  cvt(V_w, Vb,  (size_t)O * H,     stream);
  cvt(h0,  h0b, (size_t)B * H,     stream);
  zero_bar_kernel<<<1, 1, 0, stream>>>(bar);

  dim3 blk(128);

  // Phase 1: wx = x * W_w^T + (W_b + b). M=B*S, K=E, N=H.
  gemm_wmma_kernel<0><<<dim3((B * S) / 64, H / 128), blk, 0, stream>>>(
      xb, Wb, W_b, bb, wx, B * S, H, E);

  // Phase 2: persistent scan, one launch, 16 WGs, 128KB dynamic LDS each.
  rnn_scan_kernel<<<dim3(NWG_SCAN), blk, 64 * H * sizeof(__bf16), stream>>>(
      h0b, Ub, U_b, wx, hs, bar);

  // Phase 3: y[b,s,:] = hs[s,b,:] * V_w^T + V_b. M=S*B, K=H, N=O.
  gemm_wmma_kernel<2><<<dim3((S * B) / 64, O / 128), blk, 0, stream>>>(
      hs, Vb, V_b, nullptr, y, S * B, O, H);
}
